// MambaBlock_83253646066059
// MI455X (gfx1250) — compile-verified, matrix operand order checked by reading, not run on hardware
//
#include <hip/hip_runtime.h>
#include <hip/hip_bf16.h>
#include <math.h>

// ---------------------------------------------------------------------------
// MambaBlock forward for MI455X (gfx1250), bf16 WMMA throughout.
// B=2 L=4096 D=2048 H=16 HD=128 BLK=64  -> M = B*L = 8192 tokens.
// ---------------------------------------------------------------------------

typedef __attribute__((ext_vector_type(16))) __bf16 v16bf;
typedef __attribute__((ext_vector_type(8)))  float  v8f;
typedef __attribute__((ext_vector_type(4)))  __bf16 v4bf;

__device__ __forceinline__ v8f wmma_bf16(v16bf a, v16bf b, v8f c) {
  return __builtin_amdgcn_wmma_f32_16x16x32_bf16(
      false, a, false, b, (short)0, c, false, false);
}

#define D_ 2048
#define H_ 16
#define HD_ 128
#define BLK_ 64
#define NC_ 64        // chunks per sequence (4096/64)

// ------------------------------- casts -------------------------------------
__global__ void cast_f32_to_bf16(const float* __restrict__ in,
                                 __bf16* __restrict__ out, int n) {
  int i = (blockIdx.x * blockDim.x + threadIdx.x) * 4;
  if (i + 3 < n) {
    float4 v = *(const float4*)(in + i);
    v4bf o;
    o[0] = (__bf16)v.x; o[1] = (__bf16)v.y;
    o[2] = (__bf16)v.z; o[3] = (__bf16)v.w;
    *(v4bf*)(out + i) = o;
  } else {
    for (; i < n; ++i) out[i] = (__bf16)in[i];
  }
}

// --------------------------- generic bf16 NT GEMM ---------------------------
// C[M x N] = A[M x K] @ B[N x K]^T. A,B bf16 row-major (K contiguous both).
// Block: 256 threads = 8 waves; block tile 128(M) x 128(N).
// Wave grid 4(M) x 2(N): each wave owns 32 rows x 64 cols = 8 accumulators.
// Register double-buffered K loop: prefetch k+32 fragments before the 8 WMMAs.
__global__ __launch_bounds__(256) void gemm_bf16_nt(
    const __bf16* __restrict__ A, const __bf16* __restrict__ Bm,
    float* __restrict__ Cf, __bf16* __restrict__ Cb, int M, int N, int K) {
  const int lane = threadIdx.x & 31;
  const int wave = threadIdx.x >> 5;
  const int wm = wave & 3;      // M sub-tile (32 rows each)
  const int wn = wave >> 2;     // N sub-tile (64 cols each)
  const int half = lane >> 4;
  const int l16  = lane & 15;
  const size_t mBase = (size_t)blockIdx.y * 128 + wm * 32;
  const int    nBase = blockIdx.x * 128 + wn * 64;

  const __bf16* arow0 = A + (mBase + l16) * K + half * 16;
  const __bf16* arow1 = A + (mBase + 16 + l16) * K + half * 16;
  const __bf16* brow0 = Bm + (size_t)(nBase +  0 + l16) * K + half * 16;
  const __bf16* brow1 = Bm + (size_t)(nBase + 16 + l16) * K + half * 16;
  const __bf16* brow2 = Bm + (size_t)(nBase + 32 + l16) * K + half * 16;
  const __bf16* brow3 = Bm + (size_t)(nBase + 48 + l16) * K + half * 16;

  v8f acc[2][4];
#pragma unroll
  for (int i = 0; i < 2; ++i)
#pragma unroll
    for (int j = 0; j < 4; ++j) acc[i][j] = (v8f){};

  v16bf a0 = *(const v16bf*)(arow0);
  v16bf a1 = *(const v16bf*)(arow1);
  v16bf b0 = *(const v16bf*)(brow0);
  v16bf b1 = *(const v16bf*)(brow1);
  v16bf b2 = *(const v16bf*)(brow2);
  v16bf b3 = *(const v16bf*)(brow3);

  for (int k0 = 0; k0 < K - 32; k0 += 32) {
    // prefetch next K-step while current WMMAs run
    v16bf a0n = *(const v16bf*)(arow0 + k0 + 32);
    v16bf a1n = *(const v16bf*)(arow1 + k0 + 32);
    v16bf b0n = *(const v16bf*)(brow0 + k0 + 32);
    v16bf b1n = *(const v16bf*)(brow1 + k0 + 32);
    v16bf b2n = *(const v16bf*)(brow2 + k0 + 32);
    v16bf b3n = *(const v16bf*)(brow3 + k0 + 32);

    acc[0][0] = wmma_bf16(a0, b0, acc[0][0]);
    acc[0][1] = wmma_bf16(a0, b1, acc[0][1]);
    acc[0][2] = wmma_bf16(a0, b2, acc[0][2]);
    acc[0][3] = wmma_bf16(a0, b3, acc[0][3]);
    acc[1][0] = wmma_bf16(a1, b0, acc[1][0]);
    acc[1][1] = wmma_bf16(a1, b1, acc[1][1]);
    acc[1][2] = wmma_bf16(a1, b2, acc[1][2]);
    acc[1][3] = wmma_bf16(a1, b3, acc[1][3]);

    a0 = a0n; a1 = a1n; b0 = b0n; b1 = b1n; b2 = b2n; b3 = b3n;
  }
  acc[0][0] = wmma_bf16(a0, b0, acc[0][0]);
  acc[0][1] = wmma_bf16(a0, b1, acc[0][1]);
  acc[0][2] = wmma_bf16(a0, b2, acc[0][2]);
  acc[0][3] = wmma_bf16(a0, b3, acc[0][3]);
  acc[1][0] = wmma_bf16(a1, b0, acc[1][0]);
  acc[1][1] = wmma_bf16(a1, b1, acc[1][1]);
  acc[1][2] = wmma_bf16(a1, b2, acc[1][2]);
  acc[1][3] = wmma_bf16(a1, b3, acc[1][3]);

#pragma unroll
  for (int i = 0; i < 2; ++i) {
#pragma unroll
    for (int nt = 0; nt < 4; ++nt) {
      int n = nBase + nt * 16 + l16;
#pragma unroll
      for (int r = 0; r < 8; ++r) {
        size_t m = mBase + i * 16 + r + 8 * half;
        if (Cf) Cf[m * N + n] = acc[i][nt][r];
        else    Cb[m * N + n] = (__bf16)acc[i][nt][r];
      }
    }
  }
}

// ------------------------------ dt = softplus -------------------------------
__global__ void dt_kernel(const __bf16* __restrict__ xb,
                          const __bf16* __restrict__ wdt,
                          const float* __restrict__ bias,
                          float* __restrict__ dtbuf, int Mtot) {
  int idx = blockIdx.x * blockDim.x + threadIdx.x;  // (m, h)
  if (idx >= Mtot * H_) return;
  int m = idx >> 4, h = idx & 15;
  const __bf16* xr = xb + (size_t)m * D_;
  const __bf16* wr = wdt + (size_t)h * D_;
  float s = bias[h];
  for (int k = 0; k < D_; ++k) s += (float)xr[k] * (float)wr[k];
  dtbuf[idx] = (s > 20.f) ? s : logf(1.f + expf(s));
}

// --------------------------------- RoPE -------------------------------------
__global__ void rope_kernel(__bf16* __restrict__ Q, __bf16* __restrict__ Kk,
                            const float* __restrict__ cosb,
                            const float* __restrict__ sinb, int Mtot, int L) {
  int idx = blockIdx.x * blockDim.x + threadIdx.x;  // (m, h, j<64)
  if (idx >= Mtot * H_ * (HD_ / 2)) return;
  int j = idx & 63, h = (idx >> 6) & 15, m = idx >> 10;
  int l = m % L;
  float c0 = cosb[l * HD_ + j],      s0 = sinb[l * HD_ + j];
  float c1 = cosb[l * HD_ + j + 64], s1 = sinb[l * HD_ + j + 64];
  size_t base = (size_t)m * D_ + h * HD_ + j;
  float q0 = (float)Q[base], q1 = (float)Q[base + 64];
  Q[base]      = (__bf16)(q0 * c0 - q1 * s0);
  Q[base + 64] = (__bf16)(q1 * c1 + q0 * s1);
  float k0 = (float)Kk[base], k1 = (float)Kk[base + 64];
  Kk[base]      = (__bf16)(k0 * c0 - k1 * s0);
  Kk[base + 64] = (__bf16)(k1 * c1 + k0 * s1);
}

// ------------------------------- X = V * dt ---------------------------------
__global__ void scale_x_kernel(__bf16* __restrict__ X,
                               const float* __restrict__ dtbuf, int n) {
  int idx = blockIdx.x * blockDim.x + threadIdx.x;
  if (idx >= n) return;
  int m = idx >> 11;           // /2048
  int h = (idx & 2047) >> 7;   // col/128
  X[idx] = (__bf16)((float)X[idx] * dtbuf[m * H_ + h]);
}

// ------------------------ SSD per-chunk (Y_diag + states) -------------------
// One block per (b, h, chunk). 256 threads = 8 waves.
__global__ __launch_bounds__(256) void ssd_chunk_kernel(
    const __bf16* __restrict__ Qb,   // C  (m, D)
    const __bf16* __restrict__ Kb,   // B  (m, D)
    const __bf16* __restrict__ Xb,   // X  (m, D)
    const float* __restrict__ dtbuf, const float* __restrict__ A_log,
    float* __restrict__ yacc, __bf16* __restrict__ cstates, int L) {
  __shared__ __bf16 lC[64][128];    // C[t][n]
  __shared__ __bf16 lB[64][128];    // B[s][n]
  __shared__ __bf16 lBT[128][64];   // B^T[n][s]
  __shared__ __bf16 lXT[128][64];   // X^T[p][s]
  __shared__ __bf16 lXD[128][64];   // (X*decay)^T[p][s]
  __shared__ __bf16 lG[64][64];     // masked C B^T
  __shared__ float la[64], lacs[64], ldec[64];

  const int blk = blockIdx.x;
  const int c = blk % NC_;
  const int bh = blk / NC_;
  const int h = bh & 15, b = bh >> 4;
  const int m0 = b * L + c * BLK_;
  const int tid = threadIdx.x, lane = tid & 31, wave = tid >> 5;
  const int half = lane >> 4, l16 = lane & 15;

  // cumsum of a first, so the decay can be folded into the staged X tile
  if (tid < 64) la[tid] = -expf(A_log[h]) * dtbuf[(m0 + tid) * H_ + h];
  __syncthreads();
  if (tid == 0) {
    float s = 0.f;
    for (int t = 0; t < 64; ++t) { s += la[t]; lacs[t] = s; }
  }
  __syncthreads();
  if (tid < 64) ldec[tid] = expf(lacs[63] - lacs[tid]);
  __syncthreads();

  for (int i = tid; i < 64 * 128; i += 256) {
    int t = i >> 7, d = i & 127;
    size_t g = (size_t)(m0 + t) * D_ + h * HD_ + d;
    __bf16 bv = Kb[g];
    __bf16 xv = Xb[g];
    lC[t][d]  = Qb[g];
    lB[t][d]  = bv;
    lBT[d][t] = bv;
    lXT[d][t] = xv;
    lXD[d][t] = (__bf16)((float)xv * ldec[t]);
  }
  __syncthreads();

  // ---- G = C @ B^T (64x64, K=128), then mask with exp(segsum) ----
#pragma unroll
  for (int q = 0; q < 2; ++q) {
    int tIdx = wave * 2 + q, ti = tIdx >> 2, tj = tIdx & 3;
    v8f acc = (v8f){};
#pragma unroll
    for (int k0 = 0; k0 < 128; k0 += 32) {
      v16bf a  = *(const v16bf*)(&lC[ti * 16 + l16][k0 + half * 16]);
      v16bf bf = *(const v16bf*)(&lB[tj * 16 + l16][k0 + half * 16]);
      acc = wmma_bf16(a, bf, acc);
    }
#pragma unroll
    for (int r = 0; r < 8; ++r) {
      int t = ti * 16 + r + 8 * half;
      int s = tj * 16 + l16;
      float v = (s <= t) ? acc[r] * expf(lacs[t] - lacs[s]) : 0.f;
      lG[t][s] = (__bf16)v;
    }
  }
  __syncthreads();  // lG visible

  // ---- Y_diag = Gm @ X (64x128, K=64) -> yacc ----
#pragma unroll
  for (int q = 0; q < 4; ++q) {
    int tIdx = wave * 4 + q, ti = tIdx >> 3, tj = tIdx & 7;
    v8f acc = (v8f){};
#pragma unroll
    for (int k0 = 0; k0 < 64; k0 += 32) {
      v16bf a  = *(const v16bf*)(&lG[ti * 16 + l16][k0 + half * 16]);
      v16bf bf = *(const v16bf*)(&lXT[tj * 16 + l16][k0 + half * 16]);
      acc = wmma_bf16(a, bf, acc);
    }
#pragma unroll
    for (int r = 0; r < 8; ++r) {
      int t = ti * 16 + r + 8 * half;
      int p = tj * 16 + l16;
      yacc[(size_t)(m0 + t) * D_ + h * HD_ + p] = acc[r];
    }
  }

  // ---- chunk state = (X*decay)^T @ B  (128x128, K=64) -> cstates ----
  const size_t cbase = ((size_t)bh * NC_ + c) * (HD_ * HD_);
#pragma unroll
  for (int q = 0; q < 8; ++q) {
    int tIdx = wave * 8 + q, ti = tIdx >> 3, tj = tIdx & 7;
    v8f acc = (v8f){};
#pragma unroll
    for (int k0 = 0; k0 < 64; k0 += 32) {
      v16bf a  = *(const v16bf*)(&lXD[ti * 16 + l16][k0 + half * 16]);
      v16bf bf = *(const v16bf*)(&lBT[tj * 16 + l16][k0 + half * 16]);
      acc = wmma_bf16(a, bf, acc);
    }
#pragma unroll
    for (int r = 0; r < 8; ++r) {
      int p = ti * 16 + r + 8 * half;
      int n = tj * 16 + l16;
      cstates[cbase + (size_t)p * HD_ + n] = (__bf16)acc[r];
    }
  }
}

// ------------------- chunk scan + Y_off (one block per (b,h)) ---------------
__global__ __launch_bounds__(256) void scan_kernel(
    const __bf16* __restrict__ Qb, const float* __restrict__ dtbuf,
    const float* __restrict__ A_log, const __bf16* __restrict__ cstates,
    float* __restrict__ yacc, int L) {
  __shared__ float  S [128][128];   // running state, fp32 (64 KB)
  __shared__ __bf16 Sb[128][128];   // bf16 mirror for WMMA operand (32 KB)
  __shared__ __bf16 lCe[64][128];   // C * exp(A_cs)
  __shared__ float la[64], lacs[64];

  const int bh = blockIdx.x;
  const int h = bh & 15, b = bh >> 4;
  const int tid = threadIdx.x, lane = tid & 31, wave = tid >> 5;
  const int half = lane >> 4, l16 = lane & 15;
  const float aneg = -expf(A_log[h]);

  for (int i = tid; i < 128 * 128; i += 256) {
    (&S[0][0])[i] = 0.f;
    (&Sb[0][0])[i] = (__bf16)0.f;
  }

  for (int c = 0; c < NC_; ++c) {
    const int m0 = b * L + c * BLK_;
    if (tid < 64) la[tid] = aneg * dtbuf[(m0 + tid) * H_ + h];
    __syncthreads();
    if (tid == 0) {
      float s = 0.f;
      for (int t = 0; t < 64; ++t) { s += la[t]; lacs[t] = s; }
    }
    __syncthreads();
    for (int i = tid; i < 64 * 128; i += 256) {
      int t = i >> 7, n = i & 127;
      lCe[t][n] = (__bf16)((float)Qb[(size_t)(m0 + t) * D_ + h * HD_ + n] *
                           expf(lacs[t]));
    }
    __syncthreads();

    // Y_off[t,p] = sum_n Ce[t,n] * S[p][n]  (64x128, K=128); yacc += Y_off
#pragma unroll
    for (int q = 0; q < 4; ++q) {
      int tIdx = wave * 4 + q, ti = tIdx >> 3, tj = tIdx & 7;
      v8f acc = (v8f){};
      int p = tj * 16 + l16;
#pragma unroll
      for (int k0 = 0; k0 < 128; k0 += 32) {
        v16bf a  = *(const v16bf*)(&lCe[ti * 16 + l16][k0 + half * 16]);
        v16bf bf = *(const v16bf*)(&Sb[p][k0 + half * 16]);
        acc = wmma_bf16(a, bf, acc);
      }
#pragma unroll
      for (int r = 0; r < 8; ++r) {
        int t = ti * 16 + r + 8 * half;
        size_t off = (size_t)(m0 + t) * D_ + h * HD_ + tj * 16 + l16;
        yacc[off] += acc[r];
      }
    }
    __syncthreads();  // all reads of S/Sb done

    // S = S * exp(Asum_chunk) + cstate_c ; refresh bf16 mirror
    const float dTot = expf(lacs[63]);
    const size_t cbase = ((size_t)bh * NC_ + c) * (HD_ * HD_);
    for (int i = tid; i < 128 * 128; i += 256) {
      float v = (&S[0][0])[i] * dTot + (float)cstates[cbase + i];
      (&S[0][0])[i] = v;
      (&Sb[0][0])[i] = (__bf16)v;
    }
    __syncthreads();
  }
}

// ------------------------------- host side ----------------------------------
static inline size_t alignup(size_t x) { return (x + 255) & ~(size_t)255; }

extern "C" void kernel_launch(void* const* d_in, const int* in_sizes, int n_in,
                              void* d_out, int out_size, void* d_ws,
                              size_t ws_size, hipStream_t stream) {
  (void)in_sizes; (void)n_in; (void)out_size; (void)ws_size;
  const int B = 2, L = 4096, MT = B * L;  // 8192 tokens

  const float* x    = (const float*)d_in[0];
  const float* cosb = (const float*)d_in[1];
  const float* sinb = (const float*)d_in[2];
  const float* wq   = (const float*)d_in[3];
  const float* wk   = (const float*)d_in[4];
  const float* wv   = (const float*)d_in[5];
  const float* wo   = (const float*)d_in[6];
  const float* wdtw = (const float*)d_in[7];
  const float* wdtb = (const float*)d_in[8];
  const float* alog = (const float*)d_in[9];
  float* out = (float*)d_out;

  // workspace carve-up
  char* w = (char*)d_ws;
  __bf16* xb  = (__bf16*)w; w += alignup((size_t)MT * D_ * 2);
  __bf16* wqb = (__bf16*)w; w += alignup((size_t)D_ * D_ * 2);
  __bf16* wkb = (__bf16*)w; w += alignup((size_t)D_ * D_ * 2);
  __bf16* wvb = (__bf16*)w; w += alignup((size_t)D_ * D_ * 2);
  __bf16* wob = (__bf16*)w; w += alignup((size_t)D_ * D_ * 2);
  __bf16* wdt = (__bf16*)w; w += alignup((size_t)H_ * D_ * 2);
  __bf16* Qb  = (__bf16*)w; w += alignup((size_t)MT * D_ * 2);
  __bf16* Kb  = (__bf16*)w; w += alignup((size_t)MT * D_ * 2);
  __bf16* Xb  = (__bf16*)w; w += alignup((size_t)MT * D_ * 2);
  float*  dtb = (float*)w;  w += alignup((size_t)MT * H_ * 4);
  float*  yac = (float*)w;  w += alignup((size_t)MT * D_ * 4);
  __bf16* cst = (__bf16*)w; w += alignup((size_t)B * H_ * NC_ * HD_ * HD_ * 2);
  __bf16* yb  = (__bf16*)w; w += alignup((size_t)MT * D_ * 2);

  const int TB = 256;
  // 1) fp32 -> bf16 casts (4 elems/thread)
  cast_f32_to_bf16<<<(MT * D_ / 4 + TB - 1) / TB, TB, 0, stream>>>(x, xb, MT * D_);
  cast_f32_to_bf16<<<(D_ * D_ / 4 + TB - 1) / TB, TB, 0, stream>>>(wq, wqb, D_ * D_);
  cast_f32_to_bf16<<<(D_ * D_ / 4 + TB - 1) / TB, TB, 0, stream>>>(wk, wkb, D_ * D_);
  cast_f32_to_bf16<<<(D_ * D_ / 4 + TB - 1) / TB, TB, 0, stream>>>(wv, wvb, D_ * D_);
  cast_f32_to_bf16<<<(D_ * D_ / 4 + TB - 1) / TB, TB, 0, stream>>>(wo, wob, D_ * D_);
  cast_f32_to_bf16<<<(H_ * D_ / 4 + TB - 1) / TB, TB, 0, stream>>>(wdtw, wdt, H_ * D_);

  // 2) projections: Q/K/V = x @ W^T  (bf16 WMMA, bf16 out)
  dim3 ggrid(D_ / 128, MT / 128);
  gemm_bf16_nt<<<ggrid, TB, 0, stream>>>(xb, wqb, nullptr, Qb, MT, D_, D_);
  gemm_bf16_nt<<<ggrid, TB, 0, stream>>>(xb, wkb, nullptr, Kb, MT, D_, D_);
  gemm_bf16_nt<<<ggrid, TB, 0, stream>>>(xb, wvb, nullptr, Xb, MT, D_, D_);

  // 3) dt = softplus(x @ wdt^T + b)
  dt_kernel<<<(MT * H_ + TB - 1) / TB, TB, 0, stream>>>(xb, wdt, wdtb, dtb, MT);

  // 4) RoPE on Q, K (in place); X = V * dt
  rope_kernel<<<(MT * H_ * (HD_ / 2) + TB - 1) / TB, TB, 0, stream>>>(
      Qb, Kb, cosb, sinb, MT, L);
  scale_x_kernel<<<(MT * D_ + TB - 1) / TB, TB, 0, stream>>>(Xb, dtb, MT * D_);

  // 5) SSD chunk-local work: Y_diag + per-chunk states
  ssd_chunk_kernel<<<B * H_ * NC_, TB, 0, stream>>>(Qb, Kb, Xb, dtb, alog, yac,
                                                    cst, L);

  // 6) sequential chunk scan + Y_off accumulation
  scan_kernel<<<B * H_, TB, 0, stream>>>(Qb, dtb, alog, cst, yac, L);

  // 7) out = y @ wo^T  (fp32 out to d_out)
  cast_f32_to_bf16<<<(MT * D_ / 4 + TB - 1) / TB, TB, 0, stream>>>(yac, yb, MT * D_);
  gemm_bf16_nt<<<ggrid, TB, 0, stream>>>(yb, wob, out, nullptr, MT, D_, D_);
}
